// Attention_77790447665699
// MI455X (gfx1250) — matrix operand order repaired and checked by reading, not compiled, not run on hardware
//
#include <hip/hip_runtime.h>

// ---------------------------------------------------------------------------
// Causal attention forward (flash-attention style) for gfx1250 (MI455X).
// B=16, N=2048, D=256, fp32 in/out. Math path: bf16 WMMA (v_wmma_f32_16x16x32_bf16)
// with f32 accumulation and exp2-based online softmax.
// ---------------------------------------------------------------------------

namespace {
constexpr int kBatch   = 16;
constexpr int kSeq     = 2048;
constexpr int kD       = 256;
constexpr int kBM      = 128;            // query rows per workgroup
constexpr int kBN      = 32;             // key rows per tile
constexpr int kWaves   = kBM / 16;       // 8 waves, 16 query rows each
constexpr int kThreads = kWaves * 32;    // 256 threads (wave32)
constexpr int kDSteps  = kD / 32;        // 8  K-steps of 32 for QK^T
constexpr int kCBlk    = kD / 16;        // 16 column blocks of D for O
constexpr float kNegBig = -3.0e38f;      // effectively -inf, avoids inf-inf
// softmax scale folded with log2(e): S' = (q.k)/sqrt(D) * log2(e)
constexpr float kQScale = 0.0625f * 1.4426950408889634f;
}

typedef __bf16 bf16_t;
typedef bf16_t v16bf __attribute__((ext_vector_type(16)));
typedef bf16_t v8bf  __attribute__((ext_vector_type(8)));
typedef float  v8f   __attribute__((ext_vector_type(8)));

__device__ __forceinline__ float rowred_max16(float v) {
  // reduce across the 16 lanes of each half-wave (xor masks stay in-half)
  v = fmaxf(v, __shfl_xor(v, 1, 32));
  v = fmaxf(v, __shfl_xor(v, 2, 32));
  v = fmaxf(v, __shfl_xor(v, 4, 32));
  v = fmaxf(v, __shfl_xor(v, 8, 32));
  return v;
}

__device__ __forceinline__ float rowred_sum16(float v) {
  v += __shfl_xor(v, 1, 32);
  v += __shfl_xor(v, 2, 32);
  v += __shfl_xor(v, 4, 32);
  v += __shfl_xor(v, 8, 32);
  return v;
}

__global__ __launch_bounds__(kThreads)
void flash_attn_fwd(const float* __restrict__ Q, const float* __restrict__ K,
                    const float* __restrict__ V, const int* __restrict__ pmask,
                    float* __restrict__ O) {
  // LDS staging: K tile as bf16 row-major [key][d], V tile transposed [d][key],
  // plus a per-wave scratch to transpose P from C-layout into A-layout.
  __shared__ alignas(64) bf16_t Kl[kBN * kD];          // 16 KB
  __shared__ alignas(64) bf16_t VT[kD * kBN];          // 16 KB
  __shared__ alignas(64) bf16_t Pl[kWaves * 16 * kBN]; // 8 KB

  const int b     = blockIdx.y;
  const int q0    = blockIdx.x * kBM;
  const int tid   = threadIdx.x;
  const int wave  = tid >> 5;
  const int lane  = tid & 31;
  const int laneN = lane & 15;   // column index within a 16-wide tile
  const int laneH = lane >> 4;   // which half of the wave

  const float* Qb = Q + (size_t)b * kSeq * kD;
  const float* Kb = K + (size_t)b * kSeq * kD;
  const float* Vb = V + (size_t)b * kSeq * kD;
  const int*   pm = pmask + (size_t)b * kSeq;
  float*       Ob = O + (size_t)b * kSeq * kD;

  const int rowbase = q0 + wave * 16;       // first query row of this wave
  const int myrow   = rowbase + laneN;      // A-layout: lane L holds row L%16
  bf16_t* myP = &Pl[wave * 16 * kBN];

  // ---- Load this wave's 16xD Q slab into WMMA A-layout registers -----------
  // A 16x32 in this code's contiguous k order: lanes 0-15 hold K {0..15}, lanes 16-31 hold
  // K {16..31}; vector elems 0..15 = contiguous K (matches B loads).
  v16bf qa[kDSteps];
  {
    const float* qrow = Qb + (size_t)myrow * kD;
    const int off = laneH ? 16 : 0;
#pragma unroll
    for (int s = 0; s < kDSteps; ++s) {
      const float4 c0a = *(const float4*)(qrow + s * 32 + off);
      const float4 c0b = *(const float4*)(qrow + s * 32 + off + 4);
      const float4 c1a = *(const float4*)(qrow + s * 32 + 8 + off);
      const float4 c1b = *(const float4*)(qrow + s * 32 + 8 + off + 4);
      qa[s][0]  = (bf16_t)(c0a.x * kQScale);
      qa[s][1]  = (bf16_t)(c0a.y * kQScale);
      qa[s][2]  = (bf16_t)(c0a.z * kQScale);
      qa[s][3]  = (bf16_t)(c0a.w * kQScale);
      qa[s][4]  = (bf16_t)(c0b.x * kQScale);
      qa[s][5]  = (bf16_t)(c0b.y * kQScale);
      qa[s][6]  = (bf16_t)(c0b.z * kQScale);
      qa[s][7]  = (bf16_t)(c0b.w * kQScale);
      qa[s][8]  = (bf16_t)(c1a.x * kQScale);
      qa[s][9]  = (bf16_t)(c1a.y * kQScale);
      qa[s][10] = (bf16_t)(c1a.z * kQScale);
      qa[s][11] = (bf16_t)(c1a.w * kQScale);
      qa[s][12] = (bf16_t)(c1b.x * kQScale);
      qa[s][13] = (bf16_t)(c1b.y * kQScale);
      qa[s][14] = (bf16_t)(c1b.z * kQScale);
      qa[s][15] = (bf16_t)(c1b.w * kQScale);
    }
  }

  // ---- Online-softmax state and output accumulators ------------------------
  v8f acc[kCBlk];
#pragma unroll
  for (int c = 0; c < kCBlk; ++c)
#pragma unroll
    for (int r = 0; r < 8; ++r) acc[c][r] = 0.0f;

  float mstat[8], lstat[8];   // per-row max / sum, replicated over 16 lanes
#pragma unroll
  for (int r = 0; r < 8; ++r) { mstat[r] = kNegBig; lstat[r] = 0.0f; }

  const int ntiles = (q0 + kBM) / kBN;  // causal: keys < q0+kBM
  const int wave_last_row = rowbase + 15;

  for (int kt = 0; kt < ntiles; ++kt) {
    const int k0 = kt * kBN;

    __syncthreads();  // previous tile's LDS consumers are done

    // ---- Cooperative staging: K tile -> bf16 row-major, V tile -> bf16^T ---
#pragma unroll
    for (int i = 0; i < (kBN * kD / 4) / kThreads; ++i) {   // 8 float4 / thread
      const int f4   = i * kThreads + tid;      // 0..2047
      const int krow = f4 >> 6;                 // 64 float4 per row
      const int c4   = (f4 & 63) * 4;
      const float4 kv = *(const float4*)(Kb + (size_t)(k0 + krow) * kD + c4);
      const float4 vv = *(const float4*)(Vb + (size_t)(k0 + krow) * kD + c4);
      bf16_t* kd = &Kl[krow * kD + c4];
      kd[0] = (bf16_t)kv.x; kd[1] = (bf16_t)kv.y;
      kd[2] = (bf16_t)kv.z; kd[3] = (bf16_t)kv.w;
      VT[(c4 + 0) * kBN + krow] = (bf16_t)vv.x;
      VT[(c4 + 1) * kBN + krow] = (bf16_t)vv.y;
      VT[(c4 + 2) * kBN + krow] = (bf16_t)vv.z;
      VT[(c4 + 3) * kBN + krow] = (bf16_t)vv.w;
    }

    // Prefetch next tile (global_prefetch_b8) while this one is consumed.
    if (kt + 1 < ntiles) {
      const int nk0 = k0 + kBN;
      __builtin_prefetch(Kb + (size_t)(nk0 + lane) * kD, 0, 1);
      __builtin_prefetch(Vb + (size_t)(nk0 + lane) * kD, 0, 1);
    }

    __syncthreads();  // tile staged

    if (k0 > wave_last_row) continue;  // tile entirely in the masked future

    // ---- S = Q * K^T : two 16x16 f32 tiles (cols 0-15 and 16-31) -----------
    v8f s0, s1;
#pragma unroll
    for (int r = 0; r < 8; ++r) { s0[r] = 0.0f; s1[r] = 0.0f; }
#pragma unroll
    for (int s = 0; s < kDSteps; ++s) {
      // B-layout 32x16: lane n%16 = column (key row), lanes 0-15 hold K(d)
      // 0..15 of the step, lanes 16-31 hold 16..31 — contiguous in Kl.
      const v16bf b0 = *(const v16bf*)&Kl[laneN * kD + s * 32 + laneH * 16];
      const v16bf b1 = *(const v16bf*)&Kl[(16 + laneN) * kD + s * 32 + laneH * 16];
      s0 = __builtin_amdgcn_wmma_f32_16x16x32_bf16(false, qa[s], false, b0,
                                                   (short)0, s0, false, false);
      s1 = __builtin_amdgcn_wmma_f32_16x16x32_bf16(false, qa[s], false, b1,
                                                   (short)0, s1, false, false);
    }

    // ---- Mask + online softmax --------------------------------------------
    const int col0 = k0 + laneN;
    const int col1 = k0 + 16 + laneN;
    const bool pad0 = pm[col0] != 0;
    const bool pad1 = pm[col1] != 0;
#pragma unroll
    for (int r = 0; r < 8; ++r) {
      const int row = rowbase + r + 8 * laneH;  // C-layout row of VGPR r
      float e0 = (pad0 && col0 <= row) ? s0[r] : kNegBig;
      float e1 = (pad1 && col1 <= row) ? s1[r] : kNegBig;
      const float mx   = rowred_max16(fmaxf(e0, e1));
      const float mnew = fmaxf(mstat[r], mx);
      const float sc   = exp2f(mstat[r] - mnew);
      const float p0   = exp2f(e0 - mnew);
      const float p1   = exp2f(e1 - mnew);
      lstat[r] = lstat[r] * sc + rowred_sum16(p0 + p1);
      mstat[r] = mnew;
      s0[r] = p0; s1[r] = p1;
#pragma unroll
      for (int c = 0; c < kCBlk; ++c) acc[c][r] *= sc;
    }

    // ---- Transpose P (C-layout -> A-layout) through per-wave LDS scratch ---
#pragma unroll
    for (int r = 0; r < 8; ++r) {
      const int prow = r + 8 * laneH;
      myP[prow * kBN + laneN]      = (bf16_t)s0[r];
      myP[prow * kBN + 16 + laneN] = (bf16_t)s1[r];
    }
    // Same-wave LDS ops are in-order; wait for stores and fence the compiler.
    __asm__ volatile("s_wait_dscnt 0" ::: "memory");

    v16bf pa;
    {
      const int off = laneH ? 16 : 0;
      const v8bf lo = *(const v8bf*)&myP[laneN * kBN + off];
      const v8bf hi = *(const v8bf*)&myP[laneN * kBN + 8 + off];
#pragma unroll
      for (int j = 0; j < 8; ++j) { pa[j] = lo[j]; pa[8 + j] = hi[j]; }
    }

    // ---- O += P * V : 16 column blocks, K=32 covers the whole key tile -----
#pragma unroll
    for (int c = 0; c < kCBlk; ++c) {
      const v16bf bv = *(const v16bf*)&VT[(c * 16 + laneN) * kBN + laneH * 16];
      acc[c] = __builtin_amdgcn_wmma_f32_16x16x32_bf16(false, pa, false, bv,
                                                       (short)0, acc[c],
                                                       false, false);
    }
  }

  // ---- Epilogue: O = acc / l ----------------------------------------------
  float rinv[8];
#pragma unroll
  for (int r = 0; r < 8; ++r) rinv[r] = 1.0f / lstat[r];
#pragma unroll
  for (int c = 0; c < kCBlk; ++c) {
#pragma unroll
    for (int r = 0; r < 8; ++r) {
      const int row = rowbase + r + 8 * laneH;
      Ob[(size_t)row * kD + c * 16 + laneN] = acc[c][r] * rinv[r];
    }
  }
}

extern "C" void kernel_launch(void* const* d_in, const int* in_sizes, int n_in,
                              void* d_out, int out_size, void* d_ws, size_t ws_size,
                              hipStream_t stream) {
  (void)in_sizes; (void)n_in; (void)out_size; (void)d_ws; (void)ws_size;
  const float* Q  = (const float*)d_in[0];
  const float* K  = (const float*)d_in[1];
  const float* V  = (const float*)d_in[2];
  const int* pmsk = (const int*)d_in[3];
  float* out = (float*)d_out;

  dim3 grid(kSeq / kBM, kBatch, 1);
  dim3 block(kThreads, 1, 1);
  flash_attn_fwd<<<grid, block, 0, stream>>>(Q, K, V, pmsk, out);
}
